// Transformer_70901320122823
// MI455X (gfx1250) — matrix operand order repaired and checked by reading, not compiled, not run on hardware
//
#include <hip/hip_runtime.h>
#include <hip/hip_bf16.h>

// GPT-2-like forward for MI455X (gfx1250, wave32, WMMA).
// All GEMMs run on v_wmma_f32_16x16x32_bf16 (fp32 accumulate, bf16 operands
// converted during LDS staging). Attention is flash-style per 64x64 tile.
//
// Memory-system reasoning (MI455X: 23.3 TB/s HBM, 192 MB global L2):
//  - Weights (~600 MB fp32) dominate traffic; activations (8-32 MB) fit L2.
//  - Workgroups dispatch x-fastest, so the GEMM grid puts the M tile on x
//    (16 tiles) and the N tile on y: the 16 consecutive blocks sharing one
//    128-col weight panel (512 KB) run back-to-back -> each weight panel is
//    pulled from HBM ~once; the A panel re-reads hit L2.
//  - The 412 MB logits output is write-once -> non-temporal stores so it
//    doesn't evict the weight panels from L2.
//  - Global accesses are indexed as base[int32_offset] so the compiler keeps
//    them in the GLOBAL (not FLAT) path.

typedef __attribute__((ext_vector_type(16))) __bf16 v16bf;
typedef __attribute__((ext_vector_type(8)))  float  v8f;

__device__ __forceinline__ v8f wmma_bf16(v16bf a, v16bf b, v8f c) {
  return __builtin_amdgcn_wmma_f32_16x16x32_bf16(
      /*neg_a=*/false, a, /*neg_b=*/false, b,
      /*c_mod=*/(short)0, c, /*reuse_a=*/false, /*reuse_b=*/false);
}

// A fragment: 16x32 bf16, M-rows in LDS row-major [m][k] with given stride.
// Lane L (L<16): row M=L, K=0..15; lane L>=16: row M=L-16,
// K=16..31 (same contiguous K order as load_frag_b).
__device__ __forceinline__ v16bf load_frag_a(const __bf16* As, int mOff,
                                             int kOff, int lo, int hi,
                                             int stride) {
  const __bf16* p = As + (mOff + lo) * stride + kOff + (hi << 4);
  v16bf a;
#pragma unroll
  for (int q = 0; q < 16; ++q) a[q] = p[q];
  return a;
}

// B fragment: 32x16 bf16 (KxN), stored N-major in LDS [n][k] with stride.
// Lane L (L<16): col N=L, K=0..15; lane L>=16: col N=L-16, K=16..31.
__device__ __forceinline__ v16bf load_frag_b(const __bf16* Bs, int nOff,
                                             int kOff, int lo, int hi,
                                             int stride) {
  const __bf16* p = Bs + (nOff + lo) * stride + kOff + (hi << 4);
  v16bf b;
#pragma unroll
  for (int q = 0; q < 16; ++q) b[q] = p[q];
  return b;
}

// ---------------------------------------------------------------------------
// Embedding: x[b,t,:] = emb[idx[b,t],:] + pos[t,:]
// ---------------------------------------------------------------------------
__global__ __launch_bounds__(256) void embed_kernel(
    const int* __restrict__ idx, const float* __restrict__ emb,
    const float* __restrict__ pos, float* __restrict__ x, int T, int D) {
  int row = blockIdx.x;  // b*T + t
  int t = row % T;
  int token = idx[row];
  for (int c = threadIdx.x; c < D; c += 256)
    x[(size_t)row * D + c] =
        emb[(size_t)token * D + c] + pos[(size_t)t * D + c];
}

// ---------------------------------------------------------------------------
// LayerNorm: one block per row.
// ---------------------------------------------------------------------------
__global__ __launch_bounds__(256) void layernorm_kernel(
    const float* __restrict__ x, const float* __restrict__ w,
    const float* __restrict__ b, float* __restrict__ out, int D) {
  int row = blockIdx.x;
  const float* xr = x + (size_t)row * D;
  __shared__ float red[256];
  int tid = threadIdx.x;

  float s = 0.f;
  for (int c = tid; c < D; c += 256) s += xr[c];
  red[tid] = s;
  __syncthreads();
  for (int off = 128; off > 0; off >>= 1) {
    if (tid < off) red[tid] += red[tid + off];
    __syncthreads();
  }
  float mean = red[0] / (float)D;
  __syncthreads();

  float v = 0.f;
  for (int c = tid; c < D; c += 256) {
    float d = xr[c] - mean;
    v += d * d;
  }
  red[tid] = v;
  __syncthreads();
  for (int off = 128; off > 0; off >>= 1) {
    if (tid < off) red[tid] += red[tid + off];
    __syncthreads();
  }
  float rstd = rsqrtf(red[0] / (float)D + 1e-5f);

  for (int c = tid; c < D; c += 256)
    out[(size_t)row * D + c] = (xr[c] - mean) * rstd * w[c] + b[c];
}

// ---------------------------------------------------------------------------
// WMMA GEMM: C[M,N] = op(A[M,K] @ B[K,N] + bias) (+ resid)
// BM=128 BN=128 BK=32; 256 threads = 8 waves; wave tile 64x32 (4x2 WMMA).
// Grid: x = M tile (dispatch-fastest -> weight-panel L2 reuse), y = N tile.
// NGUARD: ragged-N (head GEMM, N=50257, odd row stride -> scalar B loads,
// guarded epilogue, non-temporal output stores).
// All flat offsets fit in int32 (max ~51.5M elements), so staging offsets are
// precomputed once and advanced by constants inside the K loop.
// ---------------------------------------------------------------------------
#define SA 34
#define SB 34

template <bool BIAS, bool GELU, bool RESID, bool NGUARD>
__global__ __launch_bounds__(256) void gemm_bf16_kernel(
    const float* __restrict__ A, const float* __restrict__ B,
    const float* __restrict__ bias, const float* __restrict__ resid,
    float* __restrict__ C, int M, int N, int K) {
  __shared__ __bf16 As[128 * SA];
  __shared__ __bf16 Bs[128 * SB];

  int tid = threadIdx.x;
  int wid = tid >> 5;
  int lane = tid & 31;
  int lo = lane & 15, hi = lane >> 4;
  int waveM = wid >> 2;  // 0..1
  int waveN = wid & 3;   // 0..3
  int relM = waveM * 64;
  int relN = waveN * 32;
  int blockRow = blockIdx.x * 128;  // x-fastest: row tiles sweep first
  int blockCol = blockIdx.y * 128;
  const int bStride = 32 * N;

  // --- Precomputed staging coordinates (int offsets, hoisted out of K loop).
  // A tile 128x32: 4 float4 per thread. id -> r = id>>3, c4 = (id&7)*4.
  int aOff[4], aDst[4];
#pragma unroll
  for (int it = 0; it < 4; ++it) {
    int id = tid + it * 256;
    int r = id >> 3;
    int c4 = (id & 7) << 2;
    aOff[it] = (blockRow + r) * K + c4;
    aDst[it] = r * SA + c4;
  }

  int bOff[16], bDst[16];
  unsigned bMask = 0;
  if (NGUARD) {
    // 16 scalar loads per thread. id -> kr = id>>7, n = id&127.
#pragma unroll
    for (int it = 0; it < 16; ++it) {
      int id = tid + it * 256;
      int kr = id >> 7;
      int n = id & 127;
      int ncol = blockCol + n;
      if (ncol < N) bMask |= (1u << it);
      bOff[it] = kr * N + (ncol < N ? ncol : 0);
      bDst[it] = n * SB + kr;
    }
  } else {
    // 4 float4 loads per thread. id -> kr = id>>5, n4 = (id&31)*4.
#pragma unroll
    for (int it = 0; it < 4; ++it) {
      int id = tid + it * 256;
      int kr = id >> 5;
      int n4 = (id & 31) << 2;
      bOff[it] = kr * N + blockCol + n4;
      bDst[it] = n4 * SB + kr;
    }
  }

  v8f acc[4][2];
#pragma unroll
  for (int i = 0; i < 4; ++i)
#pragma unroll
    for (int j = 0; j < 2; ++j) acc[i][j] = {};

  for (int k0 = 0; k0 < K; k0 += 32) {
    // Stage A (row-major [m][k]).
#pragma unroll
    for (int it = 0; it < 4; ++it) {
      float4 f = *(const float4*)(A + aOff[it]);
      aOff[it] += 32;
      __bf16* d = As + aDst[it];
      d[0] = (__bf16)f.x;
      d[1] = (__bf16)f.y;
      d[2] = (__bf16)f.z;
      d[3] = (__bf16)f.w;
    }
    // Stage B transposed into LDS as [n][k] (N-major).
    if (NGUARD) {
#pragma unroll
      for (int it = 0; it < 16; ++it) {
        float f = (bMask & (1u << it)) ? B[bOff[it]] : 0.f;
        bOff[it] += bStride;
        Bs[bDst[it]] = (__bf16)f;
      }
    } else {
#pragma unroll
      for (int it = 0; it < 4; ++it) {
        float4 f = *(const float4*)(B + bOff[it]);
        bOff[it] += bStride;
        __bf16* d = Bs + bDst[it];
        d[0 * SB] = (__bf16)f.x;
        d[1 * SB] = (__bf16)f.y;
        d[2 * SB] = (__bf16)f.z;
        d[3 * SB] = (__bf16)f.w;
      }
      if (k0 + 32 < K) __builtin_prefetch(B + bOff[0], 0, 1);
    }
    __syncthreads();

    v16bf af[4], bf[2];
#pragma unroll
    for (int i = 0; i < 4; ++i)
      af[i] = load_frag_a(As, relM + i * 16, 0, lo, hi, SA);
#pragma unroll
    for (int j = 0; j < 2; ++j)
      bf[j] = load_frag_b(Bs, relN + j * 16, 0, lo, hi, SB);
#pragma unroll
    for (int i = 0; i < 4; ++i)
#pragma unroll
      for (int j = 0; j < 2; ++j) acc[i][j] = wmma_bf16(af[i], bf[j], acc[i][j]);
    __syncthreads();
  }

  // Epilogue. C/D layout: lane L, VGPR g -> m = g + 8*(L>>4), n = L&15.
#pragma unroll
  for (int i = 0; i < 4; ++i) {
#pragma unroll
    for (int j = 0; j < 2; ++j) {
#pragma unroll
      for (int g = 0; g < 8; ++g) {
        int row = blockRow + relM + i * 16 + g + (hi << 3);
        int col = blockCol + relN + j * 16 + lo;
        if (!NGUARD || col < N) {
          float v = acc[i][j][g];
          if (BIAS) v += bias[col];
          if (GELU) v = 0.5f * v * (1.0f + erff(v * 0.70710678118654752f));
          if (RESID) v += resid[(size_t)row * N + col];
          if (NGUARD) {
            // Write-once logits stream: keep it out of L2 (NT hint).
            __builtin_nontemporal_store(v, &C[(size_t)row * N + col]);
          } else {
            C[(size_t)row * N + col] = v;
          }
        }
      }
    }
  }
}

// ---------------------------------------------------------------------------
// Flash attention: one block per (b, h, 64-row q-tile). hd = 64.
// S = Q K^T * (1/sqrt(D))  [reference scales by full D, not head_dim]
// online softmax across 64-col K tiles, O += P V, then x += O / l.
// ---------------------------------------------------------------------------
__global__ __launch_bounds__(256) void attention_kernel(
    const float* __restrict__ qkv, float* __restrict__ x, int Bn, int T,
    int D, int H) {
  const int HD = 64;
  __shared__ __bf16 Qs[64 * 66];
  __shared__ __bf16 Ks[64 * 66];
  __shared__ __bf16 VsT[64 * 66];  // transposed: [hd][krow]
  __shared__ __bf16 Ps[64 * 66];
  __shared__ float Ss[64 * 64];
  __shared__ float rowM[64], rowL[64], rowScale[64];

  int numQT = T / 64;
  int qt = blockIdx.x % numQT;
  int h = (blockIdx.x / numQT) % H;
  int b = blockIdx.x / (numQT * H);

  int tid = threadIdx.x;
  int wid = tid >> 5;
  int lane = tid & 31;
  int lo = lane & 15, hi = lane >> 4;
  int tm = wid >> 1;         // 0..3 : 16-row strip
  int tn0 = (wid & 1) * 2;   // 0 or 2 : two 16-col tiles

  const int row3 = 3 * D;
  const float inv_scale = rsqrtf((float)D);

  // Per-thread staging coordinates (constant across tiles).
  int sr = tid >> 6;          // base row (stride 4 below)
  int sc = tid & 63;          // column

  // Load Q tile.
#pragma unroll
  for (int it = 0; it < 16; ++it) {
    int r = sr + it * 4;
    float qv = qkv[(b * T + qt * 64 + r) * row3 + h * HD + sc];
    Qs[r * 66 + sc] = (__bf16)qv;
  }
  if (tid < 64) {
    rowM[tid] = -__builtin_inff();
    rowL[tid] = 0.f;
  }
  v8f oacc[2];
  oacc[0] = {};
  oacc[1] = {};

  // Hoisted K/V staging offsets as int32 (qkv is 6.3M floats; fits easily).
  // Indexing qkv[int] keeps the compiler on the GLOBAL (not FLAT) load path.
  int kvOff[16];
#pragma unroll
  for (int it = 0; it < 16; ++it) {
    int r = sr + it * 4;
    kvOff[it] = (b * T + r) * row3 + h * HD + sc;
  }
  const int kvStep = 64 * row3;
  __syncthreads();

  for (int kt = 0; kt <= qt; ++kt) {
    // Stage K (row-major: [krow][hd]) and V transposed ([hd][krow]).
#pragma unroll
    for (int it = 0; it < 16; ++it) {
      int r = sr + it * 4;
      int o = kvOff[it];
      kvOff[it] = o + kvStep;
      Ks[r * 66 + sc] = (__bf16)qkv[o + D];
      VsT[sc * 66 + r] = (__bf16)qkv[o + 2 * D];
    }
    __syncthreads();

    // S tiles: A=Q(16x64 over hd), B=K^T -> N-major layout == Ks row-major.
#pragma unroll
    for (int jt = 0; jt < 2; ++jt) {
      int tn = tn0 + jt;
      v8f s = {};
#pragma unroll
      for (int kk = 0; kk < 2; ++kk) {
        v16bf a = load_frag_a(Qs, tm * 16, kk * 32, lo, hi, 66);
        v16bf bb = load_frag_b(Ks, tn * 16, kk * 32, lo, hi, 66);
        s = wmma_bf16(a, bb, s);
      }
#pragma unroll
      for (int g = 0; g < 8; ++g) {
        int mr = tm * 16 + g + (hi << 3);
        int nc = tn * 16 + lo;
        float sv = s[g] * inv_scale;
        if (kt * 64 + nc > qt * 64 + mr) sv = -__builtin_inff();
        Ss[mr * 64 + nc] = sv;
      }
    }
    __syncthreads();

    // Online softmax: thread r handles row r.
    if (tid < 64) {
      int r = tid;
      float mo = rowM[r];
      float mx = mo;
      for (int j = 0; j < 64; ++j) mx = fmaxf(mx, Ss[r * 64 + j]);
      float sc2 = __expf(mo - mx);
      float sum = 0.f;
      for (int j = 0; j < 64; ++j) {
        float p = __expf(Ss[r * 64 + j] - mx);
        Ps[r * 66 + j] = (__bf16)p;
        sum += p;
      }
      rowL[r] = rowL[r] * sc2 + sum;
      rowM[r] = mx;
      rowScale[r] = sc2;
    }
    __syncthreads();

    // Rescale O and accumulate P @ V.
#pragma unroll
    for (int jt = 0; jt < 2; ++jt) {
      int tn = tn0 + jt;
      v8f o = oacc[jt];
#pragma unroll
      for (int g = 0; g < 8; ++g) {
        int mr = tm * 16 + g + (hi << 3);
        o[g] *= rowScale[mr];
      }
#pragma unroll
      for (int kk = 0; kk < 2; ++kk) {
        v16bf a = load_frag_a(Ps, tm * 16, kk * 32, lo, hi, 66);
        v16bf bb = load_frag_b(VsT, tn * 16, kk * 32, lo, hi, 66);
        o = wmma_bf16(a, bb, o);
      }
      oacc[jt] = o;
    }
    __syncthreads();
  }

  // x += O / l   (each (b,t,d) owned by exactly one block/lane)
#pragma unroll
  for (int jt = 0; jt < 2; ++jt) {
    int tn = tn0 + jt;
#pragma unroll
    for (int g = 0; g < 8; ++g) {
      int mr = tm * 16 + g + (hi << 3);
      int nc = tn * 16 + lo;
      int tg = qt * 64 + mr;
      float val = oacc[jt][g] / rowL[mr];
      int off = (b * T + tg) * D + h * HD + nc;
      x[off] += val;
    }
  }
}

// ---------------------------------------------------------------------------
// Host orchestration.
// ---------------------------------------------------------------------------
extern "C" void kernel_launch(void* const* d_in, const int* in_sizes, int n_in,
                              void* d_out, int out_size, void* d_ws,
                              size_t ws_size, hipStream_t stream) {
  const int Lc = 8, Bc = 2, Tc = 1024, Dc = 1024, Hc = 16, Fc = 4096,
            Vc = 50257;
  const int M = Bc * Tc;  // 2048

  const int* idx = (const int*)d_in[0];
  const float* emb = (const float*)d_in[1];
  const float* pos = (const float*)d_in[2];
  const float* ln1_w = (const float*)d_in[3];
  const float* ln1_b = (const float*)d_in[4];
  const float* qkv_w = (const float*)d_in[5];
  const float* qkv_b = (const float*)d_in[6];
  const float* ln2_w = (const float*)d_in[7];
  const float* ln2_b = (const float*)d_in[8];
  const float* fc1_w = (const float*)d_in[9];
  const float* fc1_b = (const float*)d_in[10];
  const float* fc2_w = (const float*)d_in[11];
  const float* fc2_b = (const float*)d_in[12];
  const float* lnf_w = (const float*)d_in[13];
  const float* lnf_b = (const float*)d_in[14];
  const float* head_w = (const float*)d_in[15];
  const float* head_b = (const float*)d_in[16];
  float* out = (float*)d_out;

  float* x = (float*)d_ws;                        // M*D
  float* lnout = x + (size_t)M * Dc;              // M*D
  float* qkv = lnout + (size_t)M * Dc;            // M*3D
  float* h1 = qkv + (size_t)M * 3 * Dc;           // M*F

  embed_kernel<<<M, 256, 0, stream>>>(idx, emb, pos, x, Tc, Dc);

  for (int l = 0; l < Lc; ++l) {
    layernorm_kernel<<<M, 256, 0, stream>>>(x, ln1_w + l * Dc, ln1_b + l * Dc,
                                            lnout, Dc);
    dim3 g1(M / 128, 3 * Dc / 128);  // x = M tile for weight-panel L2 reuse
    gemm_bf16_kernel<true, false, false, false><<<g1, 256, 0, stream>>>(
        lnout, qkv_w + (size_t)l * Dc * 3 * Dc, qkv_b + (size_t)l * 3 * Dc,
        nullptr, qkv, M, 3 * Dc, Dc);

    attention_kernel<<<Bc * Hc * (Tc / 64), 256, 0, stream>>>(qkv, x, Bc, Tc,
                                                              Dc, Hc);

    layernorm_kernel<<<M, 256, 0, stream>>>(x, ln2_w + l * Dc, ln2_b + l * Dc,
                                            lnout, Dc);
    dim3 g2(M / 128, Fc / 128);
    gemm_bf16_kernel<true, true, false, false><<<g2, 256, 0, stream>>>(
        lnout, fc1_w + (size_t)l * Dc * Fc, fc1_b + (size_t)l * Fc, nullptr,
        h1, M, Fc, Dc);
    dim3 g3(M / 128, Dc / 128);
    gemm_bf16_kernel<true, true, true, false><<<g3, 256, 0, stream>>>(
        h1, fc2_w + (size_t)l * Fc * Dc, fc2_b + (size_t)l * Dc, x, x, M, Dc,
        Fc);
  }

  layernorm_kernel<<<M, 256, 0, stream>>>(x, lnf_w, lnf_b, lnout, Dc);
  dim3 gh(M / 128, (Vc + 127) / 128);
  gemm_bf16_kernel<true, false, false, true><<<gh, 256, 0, stream>>>(
      lnout, head_w, head_b, nullptr, out, M, Vc, Dc);
}
